// NaryTreeLSTMCell_63513976373582
// MI455X (gfx1250) — compile-verified, matrix operand order checked by reading, not run on hardware
//
#include <hip/hip_runtime.h>
#include <hip/hip_bf16.h>
#include <stdint.h>

// Problem constants (from reference)
#define E_DIM   1024
#define H_DIM   1024
#define N_AR    2
#define B_NODES 16384
#define T_SLOTS (N_AR * B_NODES)     // 32768
#define K_REAL  16384
#define KDIM    3072                 // E + N*H  (inner GEMM dim)
#define NCOLS   5120                 // 3H (iou) + N*H (f)
#define MROWS   B_NODES

typedef __bf16 bf16_t;
typedef __attribute__((ext_vector_type(16))) __bf16        v16bf;
typedef __attribute__((ext_vector_type(8)))  float         v8f;
typedef __attribute__((ext_vector_type(8)))  unsigned int  v8u;
typedef __attribute__((ext_vector_type(4)))  int           v4i;

#define AS1 __attribute__((address_space(1)))
#define AS3 __attribute__((address_space(3)))

// ---- CDNA5 async-to-LDS path (guarded; falls back to sync copy) -----------
#if defined(__has_builtin)
#  if __has_builtin(__builtin_amdgcn_global_load_async_to_lds_b128)
#    define HAVE_ASYNC_LDS 1
#  endif
#  if __has_builtin(__builtin_amdgcn_s_wait_asynccnt)
#    define WAIT_ASYNC(n) __builtin_amdgcn_s_wait_asynccnt(n)
#  endif
#endif
#ifndef HAVE_ASYNC_LDS
#  define HAVE_ASYNC_LDS 0
#endif
#ifndef WAIT_ASYNC
#  if HAVE_ASYNC_LDS
#    define WAIT_ASYNC(n) asm volatile("s_wait_asynccnt %0" ::"i"(n))
#  else
#    define WAIT_ASYNC(n) do {} while (0)
#  endif
#endif

__device__ __forceinline__ void ld16_to_lds(const unsigned short* g,
                                            unsigned short* l) {
#if HAVE_ASYNC_LDS
  __builtin_amdgcn_global_load_async_to_lds_b128(
      (AS1 v4i*)g, (AS3 v4i*)l, 0, 0);
#else
  *(uint4*)l = *(const uint4*)g;
#endif
}

__device__ __forceinline__ unsigned short f2bf(float f) {
  union { float f; unsigned u; } v; v.f = f;
  unsigned r = v.u + 0x7FFFu + ((v.u >> 16) & 1u);   // round-to-nearest-even
  return (unsigned short)(r >> 16);
}

__device__ __forceinline__ float sigf(float x) {
  return 1.0f / (1.0f + __expf(-x));
}

// ---------------------------------------------------------------------------
// 1) Build concatenated bf16 weight Wcat[5120][3072]:
//    rows [0,3072):    [ W_ioux | W_iouh ]
//    rows [3072,5120): [ W_fx[j%H] | W_fh[j] ]   (absorbs tile(fx, N))
// ---------------------------------------------------------------------------
__global__ void build_wcat(const float* __restrict__ W_ioux,
                           const float* __restrict__ W_iouh,
                           const float* __restrict__ W_fx,
                           const float* __restrict__ W_fh,
                           unsigned short* __restrict__ Wcat) {
  unsigned idx = blockIdx.x * blockDim.x + threadIdx.x;
  if (idx >= (unsigned)(NCOLS * KDIM)) return;
  unsigned row = idx / KDIM;
  unsigned col = idx - row * KDIM;
  float v;
  if (row < 3072u) {
    v = (col < 1024u) ? W_ioux[(size_t)row * 1024u + col]
                      : W_iouh[(size_t)row * 2048u + (col - 1024u)];
  } else {
    unsigned j = row - 3072u;
    v = (col < 1024u) ? W_fx[(size_t)(j & 1023u) * 1024u + col]
                      : W_fh[(size_t)j * 2048u + (col - 1024u)];
  }
  Wcat[idx] = f2bf(v);
}

// ---------------------------------------------------------------------------
// 2) Build Z[16384][3072] bf16: cols [0,1024) = x ; cols [1024,3072) default hx0
// ---------------------------------------------------------------------------
__global__ void build_z_fill(const float* __restrict__ x,
                             const float* __restrict__ hx0,
                             unsigned short* __restrict__ Z) {
  size_t idx = (size_t)blockIdx.x * blockDim.x + threadIdx.x;
  if (idx >= (size_t)MROWS * KDIM) return;
  unsigned b   = (unsigned)(idx / KDIM);
  unsigned col = (unsigned)(idx - (size_t)b * KDIM);
  float v = (col < 1024u) ? x[(size_t)b * 1024u + col]
                          : hx0[(col - 1024u) & 1023u];
  Z[idx] = f2bf(v);
}

__global__ void inv_init(int* __restrict__ inv) {
  int t = blockIdx.x * blockDim.x + threadIdx.x;
  if (t < T_SLOTS) inv[t] = -1;
}

// 3) Scatter real child h states into Z, record inverse slot->k map for c gather
__global__ void scatter_h(const float* __restrict__ h,
                          const int* __restrict__ hidden_idx,
                          unsigned short* __restrict__ Z,
                          int* __restrict__ inv) {
  size_t idx = (size_t)blockIdx.x * blockDim.x + threadIdx.x;
  if (idx >= (size_t)K_REAL * H_DIM) return;
  unsigned k = (unsigned)(idx >> 10);
  unsigned j = (unsigned)(idx & 1023u);
  int t = hidden_idx[k];
  unsigned b = (unsigned)t >> 1;
  unsigned n = (unsigned)t & 1u;
  Z[(size_t)b * KDIM + 1024u + n * 1024u + j] = f2bf(h[idx]);
  if (j == 0) inv[t] = (int)k;
}

// ---------------------------------------------------------------------------
// 4) G = Z @ Wcat^T  (M=16384, N=5120, K=3072)  bf16 WMMA, f32 accumulate
//    256 threads = 8 waves; block tile 256x128; K-step 32; double-buffered
//    async-to-LDS staging. Waves: 4(M)x2(N), wave tile 64x64 = 4x4 WMMA tiles.
// ---------------------------------------------------------------------------
#define TM   256
#define TN   128
#define KC   32
#define LROW 40              // bf16 per LDS row (80 B, conflict-avoiding pad)
#define ZELE (TM * LROW)     // 10240 ushorts per Z buffer
#define WELE (TN * LROW)     // 5120  ushorts per W buffer
#define NIT  (KDIM / KC)     // 96

__global__ __launch_bounds__(256) void gemm_bf16_wmma(
    const unsigned short* __restrict__ Z,
    const unsigned short* __restrict__ W,
    float* __restrict__ G) {
  __shared__ unsigned short zt[2 * ZELE];   // 40 KB
  __shared__ unsigned short wt[2 * WELE];   // 20 KB

  const int bm   = blockIdx.x;        // 64 blocks along M
  const int bn   = blockIdx.y;        // 40 blocks along N
  const int tid  = threadIdx.x;
  const int lane = tid & 31;
  const int wave = tid >> 5;
  const int wm   = wave & 3;          // wave's 64-row strip (4 strips)
  const int wn   = wave >> 2;         // wave's 64-col strip (2 strips)
  const int r    = lane & 15;         // row (A) / col (B) within 16
  const int kh   = lane >> 4;         // lane-half (K interleave select)

  v8f acc[4][4];
  const v8f vzero = {0.f,0.f,0.f,0.f,0.f,0.f,0.f,0.f};
  #pragma unroll
  for (int mt = 0; mt < 4; ++mt)
    #pragma unroll
    for (int nt = 0; nt < 4; ++nt)
      acc[mt][nt] = vzero;

  const unsigned short* Zg = Z + (size_t)bm * TM * KDIM;
  const unsigned short* Wg = W + (size_t)bn * TN * KDIM;

  // Stage one KC-slice into buffer `buf`: 6 x 16B chunks per thread.
  // Z: 256x32 bf16 = 1024 chunks (4/thread); W: 128x32 = 512 chunks (2/thread).
  #define STAGE(k0, buf)                                                      \
    do {                                                                      \
      unsigned short* zb = &zt[(buf) * ZELE];                                 \
      unsigned short* wb = &wt[(buf) * WELE];                                 \
      _Pragma("unroll")                                                       \
      for (int q = 0; q < 4; ++q) {                                          \
        int ch = tid + 256 * q;                                               \
        int row = ch >> 2, co = (ch & 3) * 8;                                 \
        ld16_to_lds(Zg + (size_t)row * KDIM + (k0) + co,                      \
                    zb + row * LROW + co);                                    \
      }                                                                       \
      _Pragma("unroll")                                                       \
      for (int q = 0; q < 2; ++q) {                                          \
        int ch = tid + 256 * q;                                               \
        int row = ch >> 2, co = (ch & 3) * 8;                                 \
        ld16_to_lds(Wg + (size_t)row * KDIM + (k0) + co,                      \
                    wb + row * LROW + co);                                    \
      }                                                                       \
    } while (0)

  STAGE(0, 0);

  for (int it = 0; it < NIT; ++it) {
    const int buf = it & 1;
    if (it + 1 < NIT) {
      STAGE((it + 1) * KC, buf ^ 1);   // prefetch next slice into other buffer
      WAIT_ASYNC(6);                   // 6 in-flight = next slice; current done
    } else {
      WAIT_ASYNC(0);
    }
    __syncthreads();                   // all waves' current-buffer writes visible

    const unsigned short* zb = &zt[buf * ZELE];
    const unsigned short* wb = &wt[buf * WELE];

    // A fragments (16-bit A layout: lane half kh holds K {kh*8+0..7, 16+kh*8+0..7})
    v16bf afrag[4];
    #pragma unroll
    for (int mt = 0; mt < 4; ++mt) {
      const unsigned short* p = zb + (wm * 64 + mt * 16 + r) * LROW + kh * 8;
      uint4 lo = *(const uint4*)(p);        // K = kh*8 + 0..7
      uint4 hi = *(const uint4*)(p + 16);   // K = 16 + kh*8 + 0..7
      v8u raw = {lo.x, lo.y, lo.z, lo.w, hi.x, hi.y, hi.z, hi.w};
      afrag[mt] = __builtin_bit_cast(v16bf, raw);
    }
    // B fragments (column n = lane&15; same per-lane K interleave)
    v16bf bfrag[4];
    #pragma unroll
    for (int nt = 0; nt < 4; ++nt) {
      const unsigned short* p = wb + (wn * 64 + nt * 16 + r) * LROW + kh * 8;
      uint4 lo = *(const uint4*)(p);
      uint4 hi = *(const uint4*)(p + 16);
      v8u raw = {lo.x, lo.y, lo.z, lo.w, hi.x, hi.y, hi.z, hi.w};
      bfrag[nt] = __builtin_bit_cast(v16bf, raw);
    }
    #pragma unroll
    for (int mt = 0; mt < 4; ++mt)
      #pragma unroll
      for (int nt = 0; nt < 4; ++nt)
        acc[mt][nt] = __builtin_amdgcn_wmma_f32_16x16x32_bf16(
            false, afrag[mt], false, bfrag[nt],
            (short)0, acc[mt][nt], false, false);

    __syncthreads();                   // done reading buf before it is re-staged
  }
  #undef STAGE

  // C/D layout: lane -> N = lane&15; VGPR v -> M = v + 8*(lane>>4)
  #pragma unroll
  for (int mt = 0; mt < 4; ++mt) {
    #pragma unroll
    for (int nt = 0; nt < 4; ++nt) {
      int gm = bm * TM + wm * 64 + mt * 16 + kh * 8;
      int gn = bn * TN + wn * 64 + nt * 16 + r;
      float* gp = G + (size_t)gm * NCOLS + gn;
      #pragma unroll
      for (int v = 0; v < 8; ++v)
        gp[(size_t)v * NCOLS] = acc[mt][nt][v];
    }
  }
}

// ---------------------------------------------------------------------------
// 5) Gate epilogue: add biases, activations, fc-sum over children, outputs.
// ---------------------------------------------------------------------------
__global__ void epilogue(const float* __restrict__ G,
                         const float* __restrict__ c,
                         const float* __restrict__ hx1,
                         const float* __restrict__ b_iouh,
                         const float* __restrict__ b_fh,
                         const int* __restrict__ inv,
                         float* __restrict__ out) {
  size_t idx = (size_t)blockIdx.x * blockDim.x + threadIdx.x;
  if (idx >= (size_t)MROWS * H_DIM) return;
  unsigned b = (unsigned)(idx >> 10);
  unsigned j = (unsigned)(idx & 1023u);
  const float* g = G + (size_t)b * NCOLS;
  float ip  = g[j]        + b_iouh[j];
  float op  = g[1024 + j] + b_iouh[1024 + j];
  float up  = g[2048 + j] + b_iouh[2048 + j];
  float fp0 = g[3072 + j] + b_fh[j];
  float fp1 = g[4096 + j] + b_fh[1024 + j];
  int k0 = inv[b * 2];
  int k1 = inv[b * 2 + 1];
  float c0 = (k0 >= 0) ? c[(size_t)k0 * 1024 + j] : hx1[j];
  float c1 = (k1 >= 0) ? c[(size_t)k1 * 1024 + j] : hx1[j];
  float iv = sigf(ip), ov = sigf(op), uv = tanhf(up);
  float cn = iv * uv + sigf(fp0) * c0 + sigf(fp1) * c1;
  float hn = ov * tanhf(cn);
  out[idx] = hn;                               // h_new
  out[(size_t)MROWS * H_DIM + idx] = cn;       // c_new
}

// ---------------------------------------------------------------------------
extern "C" void kernel_launch(void* const* d_in, const int* in_sizes, int n_in,
                              void* d_out, int out_size, void* d_ws, size_t ws_size,
                              hipStream_t stream) {
  const float* x      = (const float*)d_in[0];
  const float* h      = (const float*)d_in[1];
  const float* c      = (const float*)d_in[2];
  const float* hx0    = (const float*)d_in[3];
  const float* hx1    = (const float*)d_in[4];
  const float* W_ioux = (const float*)d_in[5];
  const float* W_iouh = (const float*)d_in[6];
  const float* b_iouh = (const float*)d_in[7];
  const float* W_fx   = (const float*)d_in[8];
  const float* W_fh   = (const float*)d_in[9];
  const float* b_fh   = (const float*)d_in[10];
  // d_in[11] = tree_idx: unused by the reference computation
  const int* hidden_idx = (const int*)d_in[12];
  float* out = (float*)d_out;

  char* ws = (char*)d_ws;
  size_t off = 0;
  unsigned short* Z    = (unsigned short*)(ws + off); off += (size_t)MROWS * KDIM * 2;  // 100.7 MB
  unsigned short* Wcat = (unsigned short*)(ws + off); off += (size_t)NCOLS * KDIM * 2;  //  31.5 MB
  int*            inv  = (int*)(ws + off);            off += (size_t)T_SLOTS * 4;       //   0.1 MB
  float*          G    = (float*)(ws + off);          off += (size_t)MROWS * NCOLS * 4; // 335.5 MB
  (void)ws_size; (void)in_sizes; (void)n_in; (void)out_size;

  {
    size_t n = (size_t)NCOLS * KDIM;
    build_wcat<<<dim3((unsigned)((n + 255) / 256)), dim3(256), 0, stream>>>(
        W_ioux, W_iouh, W_fx, W_fh, Wcat);
  }
  {
    size_t n = (size_t)MROWS * KDIM;
    build_z_fill<<<dim3((unsigned)((n + 255) / 256)), dim3(256), 0, stream>>>(x, hx0, Z);
  }
  inv_init<<<dim3(T_SLOTS / 256), dim3(256), 0, stream>>>(inv);
  {
    size_t n = (size_t)K_REAL * H_DIM;
    scatter_h<<<dim3((unsigned)((n + 255) / 256)), dim3(256), 0, stream>>>(
        h, hidden_idx, Z, inv);
  }
  gemm_bf16_wmma<<<dim3(MROWS / TM, NCOLS / TN), dim3(256), 0, stream>>>(Z, Wcat, G);
  {
    size_t n = (size_t)MROWS * H_DIM;
    epilogue<<<dim3((unsigned)((n + 255) / 256)), dim3(256), 0, stream>>>(
        G, c, hx1, b_iouh, b_fh, inv, out);
  }
}
